// PhysMambaModel_15023795601429
// MI455X (gfx1250) — matrix operand order repaired and checked by reading, not compiled, not run on hardware
//
#include <hip/hip_runtime.h>

// ---------------------------------------------------------------------------
// PhysMamba (bidirectional Mamba, 6 blocks) for MI455X / gfx1250.
// GEMMs: v_wmma_f32_16x16x32_bf16, 4 N-tiles per wave (16x64 strip), K double-
// buffered (2 fragment sets) so loads overlap the matrix pipe. Scan: fp32
// VALU, 64 single-wave workgroups spread over WGPs, B/C via wave shuffles.
// ---------------------------------------------------------------------------

#define B_SZ   8
#define L_SEQ  1024
#define M_ROWS (B_SZ * L_SEQ)   // 8192
#define D_MODEL 128
#define D_INNER 256
#define D_STATE 32
#define DT_RANK 8
#define N_BLOCKS 6
#define XPN     128              // x_proj padded N (72 -> 128 for 4-tile strips)

typedef __attribute__((ext_vector_type(16))) __bf16 v16bf;
typedef __attribute__((ext_vector_type(8)))  float  v8f;

__device__ inline __bf16 u2bf(unsigned short u) {
  union { unsigned short u; __bf16 b; } v; v.u = u; return v.b;
}
__device__ inline unsigned short f2bfu(float f) {
  union { float f; unsigned u; } v; v.f = f;
  unsigned r = v.u + 0x7FFFu + ((v.u >> 16) & 1u);   // round-to-nearest-even
  return (unsigned short)(r >> 16);
}

// Load one K-step's fragments: A (16x32) + 4 B tiles (32x16 each).
__device__ inline void ld_frags(const unsigned short* __restrict__ ap,
                                const unsigned short* __restrict__ wp0,
                                int K, int k0,
                                v16bf& a, v16bf& b0, v16bf& b1, v16bf& b2, v16bf& b3) {
#pragma unroll
  for (int e = 0; e < 8; ++e) {
    a[e]     = u2bf(ap[k0 + e]);
    a[8 + e] = u2bf(ap[k0 + 8 + e]);
  }
#pragma unroll
  for (int e = 0; e < 16; ++e) {
    b0[e] = u2bf(wp0[k0 + e]);
    b1[e] = u2bf(wp0[(size_t)16 * K + k0 + e]);
    b2[e] = u2bf(wp0[(size_t)32 * K + k0 + e]);
    b3[e] = u2bf(wp0[(size_t)48 * K + k0 + e]);
  }
}

// ---------------------------------------------------------------------------
// WMMA GEMM: C[M,N] (f32) = A[M,K] (bf16) * W[N,K]^T (bf16).
// One wave per 16(M) x 64(N) strip; K stepped by 64 (two double-buffered
// 32-wide sub-steps). Requires K % 64 == 0, N % 64 == 0.
// flipA: per-row time reversal within each batch (backward direction).
// ---------------------------------------------------------------------------
__global__ __launch_bounds__(256) void gemm_bf16_wmma(
    const unsigned short* __restrict__ A,
    const unsigned short* __restrict__ W,
    float* __restrict__ C,
    int M, int N, int K, int flipA)
{
  const int lane = threadIdx.x & 31;
  const int wave = blockIdx.x * (blockDim.x >> 5) + (threadIdx.x >> 5);
  const int tM = M >> 4;
  const int nGroups = N >> 6;                 // 4 tiles of 16 per wave
  if (wave >= tM * nGroups) return;           // uniform per wave
  const int tm  = wave % tM;
  const int tn0 = (wave / tM) << 2;           // first N-tile of the strip
  const int half = lane >> 4;
  const int l15  = lane & 15;

  int arow = (tm << 4) + l15;
  if (flipA) { int b = arow / L_SEQ, t = arow % L_SEQ; arow = b * L_SEQ + (L_SEQ - 1 - t); }

  const unsigned short* ap  = A + (size_t)arow * K + half * 16;
  const unsigned short* wp0 = W + (size_t)((tn0 << 4) + l15) * K + half * 16;

  v8f acc0 = {}, acc1 = {}, acc2 = {}, acc3 = {};
  v16bf aA, bA0, bA1, bA2, bA3;   // set 0
  v16bf aB, bB0, bB1, bB2, bB3;   // set 1

  ld_frags(ap, wp0, K, 0, aA, bA0, bA1, bA2, bA3);
  for (int k0 = 0; k0 < K; k0 += 64) {
    // stage set 1 (k0+32) while set 0 computes
    ld_frags(ap, wp0, K, k0 + 32, aB, bB0, bB1, bB2, bB3);
    __builtin_prefetch(ap + k0 + 128, 0, 3);     // speculative: global_prefetch_b8
    __builtin_prefetch(wp0 + k0 + 128, 0, 3);
    acc0 = __builtin_amdgcn_wmma_f32_16x16x32_bf16(false, aA, false, bA0, (short)0, acc0, false, false);
    acc1 = __builtin_amdgcn_wmma_f32_16x16x32_bf16(false, aA, false, bA1, (short)0, acc1, false, false);
    acc2 = __builtin_amdgcn_wmma_f32_16x16x32_bf16(false, aA, false, bA2, (short)0, acc2, false, false);
    acc3 = __builtin_amdgcn_wmma_f32_16x16x32_bf16(false, aA, false, bA3, (short)0, acc3, false, false);
    // stage set 0 (k0+64) while set 1 computes (last iter over-reads <=128B
    // inside the workspace; values unused)
    ld_frags(ap, wp0, K, k0 + 64, aA, bA0, bA1, bA2, bA3);
    acc0 = __builtin_amdgcn_wmma_f32_16x16x32_bf16(false, aB, false, bB0, (short)0, acc0, false, false);
    acc1 = __builtin_amdgcn_wmma_f32_16x16x32_bf16(false, aB, false, bB1, (short)0, acc1, false, false);
    acc2 = __builtin_amdgcn_wmma_f32_16x16x32_bf16(false, aB, false, bB2, (short)0, acc2, false, false);
    acc3 = __builtin_amdgcn_wmma_f32_16x16x32_bf16(false, aB, false, bB3, (short)0, acc3, false, false);
  }

  const int crow0 = (tm << 4) + (half << 3);
  float* cp = C + (size_t)crow0 * N + (tn0 << 4) + l15;
#pragma unroll
  for (int i = 0; i < 8; ++i) {
    cp[(size_t)i * N]      = acc0[i];
    cp[(size_t)i * N + 16] = acc1[i];
    cp[(size_t)i * N + 32] = acc2[i];
    cp[(size_t)i * N + 48] = acc3[i];
  }
}

// ------------------------- elementwise / small kernels ---------------------

__global__ void k_f32_to_bf16(const float* __restrict__ s, unsigned short* __restrict__ d, int n) {
  int i = blockIdx.x * blockDim.x + threadIdx.x;
  if (i < n) d[i] = f2bfu(s[i]);
}
__global__ void k_f32_to_bf16_pad(const float* __restrict__ s, unsigned short* __restrict__ d,
                                  int n_src, int n_dst) {
  int i = blockIdx.x * blockDim.x + threadIdx.x;
  if (i < n_dst) d[i] = (i < n_src) ? f2bfu(s[i]) : (unsigned short)0;
}
__global__ void k_neg_exp(const float* __restrict__ s, float* __restrict__ d, int n) {
  int i = blockIdx.x * blockDim.x + threadIdx.x;
  if (i < n) d[i] = -__expf(s[i]);
}

// pre-conv ('same', K=7, 1->128 channels) + exact GELU
__global__ void k_preconv_gelu(const float* __restrict__ x, const float* __restrict__ w,
                               const float* __restrict__ bias, float* __restrict__ h) {
  int idx = blockIdx.x * blockDim.x + threadIdx.x;        // M_ROWS*128
  if (idx >= M_ROWS * D_MODEL) return;
  int d = idx & (D_MODEL - 1); int m = idx >> 7;
  int b = m / L_SEQ, t = m % L_SEQ;
  float acc = bias[d];
#pragma unroll
  for (int k = 0; k < 7; ++k) {
    int tt = t + k - 3;
    if (tt >= 0 && tt < L_SEQ) acc += x[b * L_SEQ + tt] * w[d * 7 + k];
  }
  h[idx] = 0.5f * acc * (1.0f + erff(acc * 0.70710678f));
}

// LayerNorm over D_MODEL -> bf16
__global__ void k_layernorm(const float* __restrict__ h, const float* __restrict__ g,
                            const float* __restrict__ b, unsigned short* __restrict__ xn) {
  int m = blockIdx.x * blockDim.x + threadIdx.x;
  if (m >= M_ROWS) return;
  const float* row = h + (size_t)m * D_MODEL;
  float mu = 0.f;
  for (int d = 0; d < D_MODEL; ++d) mu += row[d];
  mu *= (1.0f / D_MODEL);
  float var = 0.f;
  for (int d = 0; d < D_MODEL; ++d) { float t = row[d] - mu; var += t * t; }
  var *= (1.0f / D_MODEL);
  float rs = rsqrtf(var + 1e-5f);
  for (int d = 0; d < D_MODEL; ++d)
    xn[(size_t)m * D_MODEL + d] = f2bfu((row[d] - mu) * rs * g[d] + b[d]);
}

// causal depthwise conv (K=8) on x = xz[:, :256], + bias, + SiLU
__global__ void k_conv_silu(const float* __restrict__ xz, const float* __restrict__ cw,
                            const float* __restrict__ cb, float* __restrict__ xc,
                            unsigned short* __restrict__ xcb) {
  int idx = blockIdx.x * blockDim.x + threadIdx.x;        // M_ROWS*256
  if (idx >= M_ROWS * D_INNER) return;
  int d = idx & (D_INNER - 1); int m = idx >> 8;
  int b = m / L_SEQ, t = m % L_SEQ;
  float acc = cb[d];
#pragma unroll
  for (int k = 0; k < 8; ++k) {
    int tt = t - 7 + k;
    if (tt >= 0) acc += xz[((size_t)(b * L_SEQ + tt)) * 512 + d] * cw[d * 8 + k];
  }
  float s = acc / (1.f + __expf(-acc));
  xc[idx] = s; xcb[idx] = f2bfu(s);
}

// dt = softplus(dbl[:, :8] @ dt_proj^T + dt_bias)   (dbl has row stride XPN)
__global__ void k_dt_softplus(const float* __restrict__ dbl, const float* __restrict__ dtw,
                              const float* __restrict__ dtb, float* __restrict__ dt) {
  int idx = blockIdx.x * blockDim.x + threadIdx.x;        // M_ROWS*256
  if (idx >= M_ROWS * D_INNER) return;
  int d = idx & (D_INNER - 1); int m = idx >> 8;
  const float* r = dbl + (size_t)m * XPN;
  float acc = dtb[d];
#pragma unroll
  for (int k = 0; k < DT_RANK; ++k) acc += r[k] * dtw[d * DT_RANK + k];
  dt[idx] = (acc > 20.f) ? acc : log1pf(__expf(acc));
}

// Selective scan fused with D-skip and SiLU(z) gating.
// Grid (B, D_INNER/32), single wave per workgroup -> spread over WGPs.
// Each lane owns one channel d; B_t/C_t broadcast via wave shuffles.
__global__ __launch_bounds__(32) void k_ssm_scan(
    const float* __restrict__ dt, const float* __restrict__ xc,
    const float* __restrict__ dbl, const float* __restrict__ xz,
    const float* __restrict__ A, const float* __restrict__ Dp,
    unsigned short* __restrict__ yg)
{
  const int b    = blockIdx.x;
  const int lane = threadIdx.x;                     // 0..31
  const int d    = blockIdx.y * 32 + lane;
  float Ar[D_STATE], h[D_STATE];
#pragma unroll
  for (int s = 0; s < D_STATE; ++s) { Ar[s] = A[d * D_STATE + s]; h[s] = 0.f; }
  const float Dd = Dp[d];
  for (int t = 0; t < L_SEQ; ++t) {
    size_t m = (size_t)b * L_SEQ + t;
    float blane = dbl[m * XPN + 8  + lane];         // B_t[lane]
    float clane = dbl[m * XPN + 40 + lane];         // C_t[lane]
    float dtv = dt[m * D_INNER + d];
    float xv  = xc[m * D_INNER + d];
    float dbu = dtv * xv;
    float acc = 0.f;
#pragma unroll
    for (int s = 0; s < D_STATE; ++s) {
      float bs = __shfl(blane, s, 32);
      float cs = __shfl(clane, s, 32);
      float dA = __expf(dtv * Ar[s]);
      h[s] = dA * h[s] + dbu * bs;
      acc += h[s] * cs;
    }
    float zv = xz[m * 512 + 256 + d];
    float g = (acc + xv * Dd) * (zv / (1.f + __expf(-zv)));
    yg[m * D_INNER + d] = f2bfu(g);
  }
}

// store out_proj result into concat buffer (bf16), optional time flip
__global__ void k_store_half(const float* __restrict__ yp, unsigned short* __restrict__ G,
                             int colOff, int flip) {
  int idx = blockIdx.x * blockDim.x + threadIdx.x;        // M_ROWS*128
  if (idx >= M_ROWS * D_MODEL) return;
  int d = idx & (D_MODEL - 1); int m = idx >> 7;
  int mo = m;
  if (flip) { int b = m / L_SEQ, t = m % L_SEQ; mo = b * L_SEQ + (L_SEQ - 1 - t); }
  G[(size_t)mo * D_INNER + colOff + d] = f2bfu(yp[idx]);
}

// residual: h += gateC + gate_b
__global__ void k_add_gate(float* __restrict__ h, const float* __restrict__ gc,
                           const float* __restrict__ gb) {
  int idx = blockIdx.x * blockDim.x + threadIdx.x;        // M_ROWS*128
  if (idx >= M_ROWS * D_MODEL) return;
  h[idx] += gc[idx] + gb[idx & (D_MODEL - 1)];
}

__global__ void k_time_mean(const float* __restrict__ h, float* __restrict__ s) {
  int b = blockIdx.x, d = threadIdx.x;                    // 8 x 128
  float acc = 0.f;
  for (int t = 0; t < L_SEQ; ++t) acc += h[((size_t)b * L_SEQ + t) * D_MODEL + d];
  s[b * D_MODEL + d] = acc * (1.0f / L_SEQ);
}

__global__ void k_se_attn(const float* __restrict__ s, const float* __restrict__ w1,
                          const float* __restrict__ b1, const float* __restrict__ w2,
                          const float* __restrict__ b2, float* __restrict__ attn) {
  __shared__ float s1[32];
  int tid = threadIdx.x;                                  // 128 threads, 1 block
  for (int b = 0; b < B_SZ; ++b) {
    __syncthreads();
    if (tid < 32) {
      float a = b1[tid];
      for (int k = 0; k < D_MODEL; ++k) a += w1[tid * D_MODEL + k] * s[b * D_MODEL + k];
      s1[tid] = fmaxf(a, 0.f);
    }
    __syncthreads();
    float a = b2[tid];
    for (int k = 0; k < 32; ++k) a += w2[tid * 32 + k] * s1[k];
    attn[b * D_MODEL + tid] = 1.f / (1.f + __expf(-a));
  }
}

__global__ void k_final(const float* __restrict__ h, const float* __restrict__ attn,
                        const float* __restrict__ ow, const float* __restrict__ ob,
                        float* __restrict__ out) {
  int m = blockIdx.x * blockDim.x + threadIdx.x;          // M_ROWS
  if (m >= M_ROWS) return;
  int b = m / L_SEQ;
  float acc = 0.f;
  for (int d = 0; d < D_MODEL; ++d)
    acc += h[(size_t)m * D_MODEL + d] * attn[b * D_MODEL + d] * ow[d];
  out[m] = acc + ob[0];
}

// ---------------------------------------------------------------------------
// host launch
// ---------------------------------------------------------------------------
static inline int cdiv(int a, int b) { return (a + b - 1) / b; }

extern "C" void kernel_launch(void* const* d_in, const int* in_sizes, int n_in,
                              void* d_out, int out_size, void* d_ws, size_t ws_size,
                              hipStream_t stream) {
  (void)in_sizes; (void)n_in; (void)out_size; (void)ws_size;
  auto F = [&](int i) { return (const float*)d_in[i]; };

  // input index map (setup_inputs dict order, nested dicts/lists in order):
  // 0:x 1:pre_conv_w 2:pre_conv_b, then per block (22 entries):
  //   +0 ln_g +1 ln_b, mf @ +2..+10, mb @ +11..+19, +20 gate_w +21 gate_b
  //   mamba params: 0 in_proj 1 conv_w 2 conv_b 3 x_proj 4 dt_proj 5 dt_bias
  //                 6 A_log 7 D 8 out_proj
  const int TAIL = 3 + N_BLOCKS * 22;   // 135: se1_w se1_b se2_w se2_b out_w out_b

  // workspace carve-up
  size_t off = 0;
  auto alloc = [&](size_t bytes) {
    void* p = (char*)d_ws + off;
    off += (bytes + 255) & ~(size_t)255;
    return p;
  };
  float*          h     = (float*)alloc((size_t)M_ROWS * 128 * 4);
  unsigned short* xn    = (unsigned short*)alloc((size_t)M_ROWS * 128 * 2);
  float*          xz    = (float*)alloc((size_t)M_ROWS * 512 * 4);
  float*          xc    = (float*)alloc((size_t)M_ROWS * 256 * 4);
  unsigned short* xcb   = (unsigned short*)alloc((size_t)M_ROWS * 256 * 2);
  float*          dbl   = (float*)alloc((size_t)M_ROWS * XPN * 4);
  float*          dt    = (float*)alloc((size_t)M_ROWS * 256 * 4);
  unsigned short* yg    = (unsigned short*)alloc((size_t)M_ROWS * 256 * 2);
  float*          yproj = (float*)alloc((size_t)M_ROWS * 128 * 4);  // also gate C
  unsigned short* Gbf   = (unsigned short*)alloc((size_t)M_ROWS * 256 * 2);
  unsigned short* wIn   = (unsigned short*)alloc((size_t)512 * 128 * 2);
  unsigned short* wX    = (unsigned short*)alloc((size_t)XPN * 256 * 2);
  unsigned short* wOut  = (unsigned short*)alloc((size_t)128 * 256 * 2);
  unsigned short* wGate = (unsigned short*)alloc((size_t)128 * 256 * 2);
  float*          Aneg  = (float*)alloc((size_t)256 * 32 * 4);
  float*          smean = (float*)alloc((size_t)B_SZ * 128 * 4);
  float*          attn  = (float*)alloc((size_t)B_SZ * 128 * 4);

  const int TB = 256;
  auto gemm = [&](const unsigned short* A, const unsigned short* W, float* C,
                  int M, int N, int K, int flip) {
    int waves = (M >> 4) * (N >> 6);     // one wave per 16x64 strip
    gemm_bf16_wmma<<<waves / 8, TB, 0, stream>>>(A, W, C, M, N, K, flip);
  };

  // stage 0: pre-conv + gelu
  k_preconv_gelu<<<cdiv(M_ROWS * 128, TB), TB, 0, stream>>>(F(0), F(1), F(2), h);

  for (int blk = 0; blk < N_BLOCKS; ++blk) {
    const int base = 3 + blk * 22;
    k_layernorm<<<cdiv(M_ROWS, TB), TB, 0, stream>>>(h, F(base + 0), F(base + 1), xn);
    k_f32_to_bf16<<<cdiv(128 * 256, TB), TB, 0, stream>>>(F(base + 20), wGate, 128 * 256);

    for (int dir = 0; dir < 2; ++dir) {
      const int p = base + 2 + dir * 9;   // mf / mb
      const int flip = dir;
      // weight prep
      k_f32_to_bf16<<<cdiv(512 * 128, TB), TB, 0, stream>>>(F(p + 0), wIn, 512 * 128);
      k_f32_to_bf16_pad<<<cdiv(XPN * 256, TB), TB, 0, stream>>>(F(p + 3), wX, 72 * 256, XPN * 256);
      k_f32_to_bf16<<<cdiv(128 * 256, TB), TB, 0, stream>>>(F(p + 8), wOut, 128 * 256);
      k_neg_exp<<<cdiv(256 * 32, TB), TB, 0, stream>>>(F(p + 6), Aneg, 256 * 32);
      // mamba pipeline
      gemm(xn, wIn, xz, M_ROWS, 512, 128, flip);                       // in_proj
      k_conv_silu<<<cdiv(M_ROWS * 256, TB), TB, 0, stream>>>(xz, F(p + 1), F(p + 2), xc, xcb);
      gemm(xcb, wX, dbl, M_ROWS, XPN, 256, 0);                         // x_proj (padded)
      k_dt_softplus<<<cdiv(M_ROWS * 256, TB), TB, 0, stream>>>(dbl, F(p + 4), F(p + 5), dt);
      k_ssm_scan<<<dim3(B_SZ, D_INNER / 32), 32, 0, stream>>>(dt, xc, dbl, xz, Aneg, F(p + 7), yg);
      gemm(yg, wOut, yproj, M_ROWS, 128, 256, 0);                      // out_proj
      k_store_half<<<cdiv(M_ROWS * 128, TB), TB, 0, stream>>>(yproj, Gbf, dir * 128, flip);
    }
    gemm(Gbf, wGate, yproj, M_ROWS, 128, 256, 0);                      // gate GEMM
    k_add_gate<<<cdiv(M_ROWS * 128, TB), TB, 0, stream>>>(h, yproj, F(base + 21));
  }

  // SE + output head
  k_time_mean<<<B_SZ, 128, 0, stream>>>(h, smean);
  k_se_attn<<<1, 128, 0, stream>>>(smean, F(TAIL + 0), F(TAIL + 1), F(TAIL + 2), F(TAIL + 3), attn);
  k_final<<<cdiv(M_ROWS, TB), TB, 0, stream>>>(h, attn, F(TAIL + 4), F(TAIL + 5), (float*)d_out);
}
